// GPT2Model_60911226192044
// MI455X (gfx1250) — compile-verified, matrix operand order checked by reading, not run on hardware
//
#include <hip/hip_runtime.h>
#include <math.h>

// ---- problem dims (GPT-2 small) ----
#define Bz  2
#define Tz  1024
#define Vz  50257
#define VzP 50304          // Vz padded to multiple of 128
#define Dz  768
#define Hz  12
#define Lz  12
#define Fz  3072
#define HDz 64

// ---- GEMM tiling ----
#define BM 128
#define BN 128
#define BK 32

typedef __attribute__((ext_vector_type(16))) __bf16 bf16x16;
typedef __attribute__((ext_vector_type(8)))  float  f32x8;

union FragU { bf16x16 v; uint4 q[2]; };

__device__ __forceinline__ unsigned short f2bf(float f) {
    union { float f; unsigned int u; } c; c.f = f;
    unsigned int u = c.u;
    unsigned int r = u + 0x7FFFu + ((u >> 16) & 1u);   // round-to-nearest-even
    return (unsigned short)(r >> 16);
}

// ================= Tensor Data Mover support =================
#if __has_builtin(__builtin_amdgcn_tensor_load_to_lds) && __has_builtin(__builtin_amdgcn_s_wait_tensorcnt)
#define USE_TDM 1
typedef __attribute__((ext_vector_type(4))) unsigned int u32x4;
typedef __attribute__((ext_vector_type(4))) int i32x4;
typedef __attribute__((ext_vector_type(8))) int i32x8;

// 2D tile load: 128 rows x BK bf16 elements, row stride = k_stride elements.
// D# per CDNA5 ISA 08_async_tensor: group0 {count=1, lds_addr, global_addr, type=2},
// group1 {data_size=2B, tensor dims, tile dims 32x128, dim0 stride}.
__device__ __forceinline__ void tdm_load_2d(unsigned lds_off, const void* gptr,
                                            unsigned k_stride)
{
    unsigned long long ga = (unsigned long long)(size_t)gptr;
    u32x4 g0;
    g0.x = 1u;                                          // count=1, user mode
    g0.y = lds_off;                                     // lds_addr (bytes)
    g0.z = (unsigned)ga;                                // global_addr[31:0]
    g0.w = (unsigned)((ga >> 32) & 0x01FFFFFFu) | (2u << 30);  // addr[56:32] | type=2
    unsigned td0 = k_stride;                            // tensor_dim0 (elements)
    unsigned td1 = 128u;                                // tensor_dim1 (rows)
    i32x8 g1;
    g1[0] = (int)(1u << 16);                            // data_size=1 -> 2 bytes
    g1[1] = (int)((td0 & 0xFFFFu) << 16);               // tensor_dim0[15:0] @ bits63:48
    g1[2] = (int)(((td0 >> 16) & 0xFFFFu) | ((td1 & 0xFFFFu) << 16)); // td0[31:16]|td1[15:0]
    g1[3] = (int)(((td1 >> 16) & 0xFFFFu) | ((unsigned)BK << 16));    // td1[31:16]|tile_dim0
    g1[4] = (int)128u;                                  // tile_dim1=128, tile_dim2=0
    g1[5] = (int)k_stride;                              // tensor_dim0_stride[31:0] (elements)
    g1[6] = 0;                                          // stride0[47:32], stride1[15:0]
    g1[7] = 0;
    i32x4 z4 = {0, 0, 0, 0};
    i32x8 z8 = {0, 0, 0, 0, 0, 0, 0, 0};
    // 6-arg variant (clang-23 / therock-10.0 headers)
    __builtin_amdgcn_tensor_load_to_lds(g0, g1, z4, z4, z8, 0);
}

__device__ __forceinline__ unsigned lds_offset(const void* p) {
    return (unsigned)(size_t)p;   // LDS flat addr: low 32 bits = LDS byte offset
}
#endif

// ============================================================
// Embedding: x[b,t,:] = wte[ids[b,t],:] + wpe[t,:]  (f32)
// ============================================================
__global__ __launch_bounds__(256)
void embed_k(const int* __restrict__ ids, const float* __restrict__ wte,
             const float* __restrict__ wpe, float* __restrict__ x)
{
    int row = blockIdx.x;              // b*T + t
    int t   = row % Tz;
    int id  = ids[row];
    for (int i = threadIdx.x; i < Dz; i += 256)
        x[(size_t)row * Dz + i] = wte[(size_t)id * Dz + i] + wpe[(size_t)t * Dz + i];
}

// ============================================================
// Weight repack: W (K,N) f32 -> Wp (Npad,K) bf16  (transpose+convert+pad)
// ============================================================
__global__ __launch_bounds__(256)
void convert_wT_k(const float* __restrict__ W, unsigned short* __restrict__ Wp,
                  int K, int N, int Npad)
{
    size_t idx = (size_t)blockIdx.x * 256 + threadIdx.x;
    if (idx >= (size_t)Npad * K) return;
    int n = (int)(idx / K), k = (int)(idx % K);
    float v = (n < N) ? W[(size_t)k * N + n] : 0.f;
    Wp[idx] = f2bf(v);
}

// W already (N,K) row-major (wte) -> Wp (Npad,K) bf16
__global__ __launch_bounds__(256)
void convert_w_k(const float* __restrict__ W, unsigned short* __restrict__ Wp,
                 int K, int N, int Npad)
{
    size_t idx = (size_t)blockIdx.x * 256 + threadIdx.x;
    if (idx >= (size_t)Npad * K) return;
    int n = (int)(idx / K), k = (int)(idx % K);
    float v = (n < N) ? W[(size_t)n * K + k] : 0.f;
    Wp[idx] = f2bf(v);
}

// ============================================================
// LayerNorm over D=768, one block (8 wave32) per row; bf16 output
// ============================================================
__global__ __launch_bounds__(256)
void layernorm_k(const float* __restrict__ x, const float* __restrict__ w,
                 const float* __restrict__ bb, unsigned short* __restrict__ y)
{
    int row = blockIdx.x;
    const float* xr = x + (size_t)row * Dz;
    int tid = threadIdx.x;
    float v0 = xr[tid], v1 = xr[tid + 256], v2 = xr[tid + 512];
    float s  = v0 + v1 + v2;
    float ss = v0 * v0 + v1 * v1 + v2 * v2;
    #pragma unroll
    for (int off = 16; off > 0; off >>= 1) {
        s  += __shfl_xor(s,  off, 32);
        ss += __shfl_xor(ss, off, 32);
    }
    __shared__ float rs[8], rss[8];
    int lane = tid & 31, wv = tid >> 5;
    if (lane == 0) { rs[wv] = s; rss[wv] = ss; }
    __syncthreads();
    if (wv == 0) {
        s  = (lane < 8) ? rs[lane]  : 0.f;
        ss = (lane < 8) ? rss[lane] : 0.f;
        #pragma unroll
        for (int off = 16; off > 0; off >>= 1) {
            s  += __shfl_xor(s,  off, 32);
            ss += __shfl_xor(ss, off, 32);
        }
        if (lane == 0) { rs[0] = s; rss[0] = ss; }
    }
    __syncthreads();
    float mean = rs[0] * (1.0f / Dz);
    float var  = rss[0] * (1.0f / Dz) - mean * mean;
    float rstd = rsqrtf(var + 1e-5f);
    y[(size_t)row * Dz + tid]       = f2bf((v0 - mean) * rstd * w[tid]       + bb[tid]);
    y[(size_t)row * Dz + tid + 256] = f2bf((v1 - mean) * rstd * w[tid + 256] + bb[tid + 256]);
    y[(size_t)row * Dz + tid + 512] = f2bf((v2 - mean) * rstd * w[tid + 512] + bb[tid + 512]);
}

// ============================================================
// WMMA GEMM: C(M,N) = act( A(M,K)bf16 @ Bp(N,K)bf16^T + bias [+ residual] )
//   flags bit0: gelu(new) epilogue
//   flags bit1: store C as bf16 (else f32)
// Double-buffered LDS; tiles fetched by the Tensor Data Mover
// (tensor_load_to_lds + s_wait_tensorcnt) when available.
// 128x128 block tile, 8 waves, each wave 4x2 accumulators of 16x16.
// ============================================================
__global__ __launch_bounds__(256)
void gemm_bf16_wmma(const unsigned short* __restrict__ A,
                    const unsigned short* __restrict__ Bp,
                    const float* __restrict__ bias,
                    const float* __restrict__ residual,
                    void* __restrict__ Cout,
                    int M, int N, int K, int flags)
{
    const bool GELU  = (flags & 1) != 0;
    const bool OUTBF = (flags & 2) != 0;

    __shared__ __align__(16) unsigned short sA[2][BM][BK];
    __shared__ __align__(16) unsigned short sB[2][BN][BK];

    int tid  = threadIdx.x;
    int lane = tid & 31, wave = tid >> 5;
    int wm   = wave & 1;        // 2 wave-rows
    int wn   = wave >> 1;       // 4 wave-cols
    int hf   = lane >> 4;       // lane half (0/1)
    int lrow = lane & 15;

    int tileM = blockIdx.y * BM;
    int tileN = blockIdx.x * BN;
    const unsigned short* Abase = A  + (size_t)tileM * K;
    const unsigned short* Bbase = Bp + (size_t)tileN * K;

    f32x8 acc[4][2] = {};
    int nk = K / BK;

#if USE_TDM
    if (tid < 32) {                       // wave 0 drives the TDM
        tdm_load_2d(lds_offset(&sA[0][0][0]), Abase, (unsigned)K);
        tdm_load_2d(lds_offset(&sB[0][0][0]), Bbase, (unsigned)K);
    }
#endif

    for (int step = 0; step < nk; ++step) {
        int buf = step & 1;
#if USE_TDM
        if (tid < 32) {
            if (step + 1 < nk) {          // issue next tile into other buffer
                tdm_load_2d(lds_offset(&sA[buf ^ 1][0][0]),
                            Abase + (size_t)(step + 1) * BK, (unsigned)K);
                tdm_load_2d(lds_offset(&sB[buf ^ 1][0][0]),
                            Bbase + (size_t)(step + 1) * BK, (unsigned)K);
                __builtin_amdgcn_s_wait_tensorcnt(2);  // current tile landed
            } else {
                __builtin_amdgcn_s_wait_tensorcnt(0);
            }
        }
        __syncthreads();
#else
        {   // manual bf16 staging, 16B per op
            const unsigned short* ga = Abase + (size_t)step * BK;
            for (int e = tid; e < (BM * BK) / 8; e += 256) {
                int m = e >> 2, c = e & 3;
                ((uint4*)&sA[buf][m][0])[c] = ((const uint4*)(ga + (size_t)m * K))[c];
            }
            const unsigned short* gb = Bbase + (size_t)step * BK;
            for (int e = tid; e < (BN * BK) / 8; e += 256) {
                int n = e >> 2, c = e & 3;
                ((uint4*)&sB[buf][n][0])[c] = ((const uint4*)(gb + (size_t)n * K))[c];
            }
        }
        __syncthreads();
#endif

        // fragment loads: 16-bit A 16x32 per-lane K packing
        // lane half 0: K {0..7, 16..23}; half 1: K {8..15, 24..31}
        FragU a[4], b[2];
        #pragma unroll
        for (int sm = 0; sm < 4; ++sm) {
            const uint4* p = (const uint4*)&sA[buf][wm * 64 + sm * 16 + lrow][0];
            a[sm].q[0] = p[hf];
            a[sm].q[1] = p[2 + hf];
        }
        #pragma unroll
        for (int sn = 0; sn < 2; ++sn) {
            const uint4* p = (const uint4*)&sB[buf][wn * 32 + sn * 16 + lrow][0];
            b[sn].q[0] = p[hf];
            b[sn].q[1] = p[2 + hf];
        }

        #pragma unroll
        for (int sm = 0; sm < 4; ++sm)
            #pragma unroll
            for (int sn = 0; sn < 2; ++sn)
                acc[sm][sn] = __builtin_amdgcn_wmma_f32_16x16x32_bf16(
                    false, a[sm].v, false, b[sn].v,
                    (short)0, acc[sm][sn], false, false);
        __syncthreads();   // keep LDS buffer safe before next overwrite
    }

    // ---- epilogue: VGPR r -> lanes0-15 M=r, lanes16-31 M=r+8 ----
    #pragma unroll
    for (int sm = 0; sm < 4; ++sm) {
        #pragma unroll
        for (int sn = 0; sn < 2; ++sn) {
            int gn = tileN + wn * 32 + sn * 16 + lrow;
            if (gn >= N) continue;
            #pragma unroll
            for (int r = 0; r < 8; ++r) {
                int gm = tileM + wm * 64 + sm * 16 + r + hf * 8;
                float v = acc[sm][sn][r];
                if (bias)     v += bias[gn];
                if (residual) v += residual[(size_t)gm * N + gn];
                if (GELU) {
                    float x3 = v * v * v;
                    v = 0.5f * v * (1.f + tanhf(0.7978845608028654f * (v + 0.044715f * x3)));
                }
                if (OUTBF) ((unsigned short*)Cout)[(size_t)gm * N + gn] = f2bf(v);
                else       ((float*)Cout)[(size_t)gm * N + gn] = v;
            }
        }
    }
}

// ============================================================
// Causal attention with online softmax; bf16 output.
// qkv: (B,T,3D) f32. One wave per query row, HD=64 -> 2 dims/lane.
// ============================================================
__global__ __launch_bounds__(256)
void attention_k(const float* __restrict__ qkv, unsigned short* __restrict__ out)
{
    int bh = blockIdx.y;
    int b = bh / Hz, h = bh % Hz;
    int lane = threadIdx.x & 31, wave = threadIdx.x >> 5;
    int qt = blockIdx.x * 8 + wave;
    const float scale = 0.125f;  // 1/sqrt(64)

    const float* qrow = qkv + ((size_t)(b * Tz + qt) * (3 * Dz)) + h * HDz;
    float q0 = qrow[lane], q1 = qrow[lane + 32];

    float mx = -INFINITY, l = 0.f, a0 = 0.f, a1 = 0.f;
    for (int j = 0; j <= qt; ++j) {
        const float* kr = qkv + ((size_t)(b * Tz + j) * (3 * Dz)) + Dz + h * HDz;
        float s = q0 * kr[lane] + q1 * kr[lane + 32];
        #pragma unroll
        for (int off = 16; off > 0; off >>= 1) s += __shfl_xor(s, off, 32);
        s *= scale;
        float mn   = fmaxf(mx, s);
        float corr = __expf(mx - mn);   // 0 on first iter
        float p    = __expf(s - mn);
        const float* vr = kr + Dz;
        a0 = a0 * corr + p * vr[lane];
        a1 = a1 * corr + p * vr[lane + 32];
        l  = l * corr + p;
        mx = mn;
    }
    float inv = 1.f / l;
    unsigned short* orow = out + ((size_t)(b * Tz + qt) * Dz) + h * HDz;
    orow[lane]      = f2bf(a0 * inv);
    orow[lane + 32] = f2bf(a1 * inv);
}

// ============================================================
// Host-side orchestration
// ============================================================
extern "C" void kernel_launch(void* const* d_in, const int* in_sizes, int n_in,
                              void* d_out, int out_size, void* d_ws, size_t ws_size,
                              hipStream_t stream)
{
    const int*   ids    = (const int*)  d_in[0];
    const float* wte    = (const float*)d_in[1];
    const float* wpe    = (const float*)d_in[2];
    const float* ln1_w  = (const float*)d_in[3];
    const float* ln1_b  = (const float*)d_in[4];
    const float* attn_w = (const float*)d_in[5];
    const float* attn_b = (const float*)d_in[6];
    const float* proj_w = (const float*)d_in[7];
    const float* proj_b = (const float*)d_in[8];
    const float* ln2_w  = (const float*)d_in[9];
    const float* ln2_b  = (const float*)d_in[10];
    const float* fc_w   = (const float*)d_in[11];
    const float* fc_b   = (const float*)d_in[12];
    const float* fc2_w  = (const float*)d_in[13];
    const float* fc2_b  = (const float*)d_in[14];
    const float* lnf_w  = (const float*)d_in[15];
    const float* lnf_b  = (const float*)d_in[16];

    const size_t BT = (size_t)Bz * Tz;   // 2048 rows
    char* ws = (char*)d_ws;
    float*          x    = (float*)ws;                       ws += BT * Dz * 4;      // residual stream f32
    unsigned short* hb   = (unsigned short*)ws;              ws += BT * Dz * 2;      // LN out bf16
    float*          qkv  = (float*)ws;                       ws += BT * 3 * Dz * 4;  // qkv f32
    unsigned short* ao   = (unsigned short*)ws;              ws += BT * Dz * 2;      // attn out bf16
    unsigned short* ff   = (unsigned short*)ws;              ws += BT * Fz * 2;      // mlp hidden bf16
    unsigned short* wbuf = (unsigned short*)ws;              ws += (size_t)Dz * Fz * 2; // packed weight scratch
    unsigned short* wtep = (unsigned short*)ws;              ws += (size_t)VzP * Dz * 2; // packed wte

    dim3 blk(256);
    const int Mrows = (int)BT;
    auto cblocks = [](size_t n) { return (unsigned)((n + 255) / 256); };

    embed_k<<<(unsigned)BT, blk, 0, stream>>>(ids, wte, wpe, x);
    // pack tied lm_head weights once: wte (V,D) -> (VzP, D) bf16
    convert_w_k<<<cblocks((size_t)VzP * Dz), blk, 0, stream>>>(wte, wtep, Dz, Vz, VzP);

    for (int l = 0; l < Lz; ++l) {
        // --- attention block ---
        layernorm_k<<<(unsigned)BT, blk, 0, stream>>>(x, ln1_w + (size_t)l * Dz,
                                                      ln1_b + (size_t)l * Dz, hb);
        convert_wT_k<<<cblocks((size_t)3 * Dz * Dz), blk, 0, stream>>>(
            attn_w + (size_t)l * Dz * 3 * Dz, wbuf, Dz, 3 * Dz, 3 * Dz);
        gemm_bf16_wmma<<<dim3((3 * Dz) / BN, Mrows / BM), blk, 0, stream>>>(
            hb, wbuf, attn_b + (size_t)l * 3 * Dz, nullptr,
            qkv, Mrows, 3 * Dz, Dz, 0);
        attention_k<<<dim3(Tz / 8, Bz * Hz), blk, 0, stream>>>(qkv, ao);
        convert_wT_k<<<cblocks((size_t)Dz * Dz), blk, 0, stream>>>(
            proj_w + (size_t)l * Dz * Dz, wbuf, Dz, Dz, Dz);
        gemm_bf16_wmma<<<dim3(Dz / BN, Mrows / BM), blk, 0, stream>>>(
            ao, wbuf, proj_b + (size_t)l * Dz, x, x, Mrows, Dz, Dz, 0);
        // --- mlp block ---
        layernorm_k<<<(unsigned)BT, blk, 0, stream>>>(x, ln2_w + (size_t)l * Dz,
                                                      ln2_b + (size_t)l * Dz, hb);
        convert_wT_k<<<cblocks((size_t)Dz * Fz), blk, 0, stream>>>(
            fc_w + (size_t)l * Dz * Fz, wbuf, Dz, Fz, Fz);
        gemm_bf16_wmma<<<dim3(Fz / BN, Mrows / BM), blk, 0, stream>>>(
            hb, wbuf, fc_b + (size_t)l * Fz, nullptr, ff, Mrows, Fz, Dz,
            1 | 2 /*gelu, bf16 out*/);
        convert_wT_k<<<cblocks((size_t)Fz * Dz), blk, 0, stream>>>(
            fc2_w + (size_t)l * Fz * Dz, wbuf, Fz, Dz, Dz);
        gemm_bf16_wmma<<<dim3(Dz / BN, Mrows / BM), blk, 0, stream>>>(
            ff, wbuf, fc2_b + (size_t)l * Dz, x, x, Mrows, Dz, Fz, 0);
    }

    // --- final LN + tied lm_head ---
    layernorm_k<<<(unsigned)BT, blk, 0, stream>>>(x, lnf_w, lnf_b, hb);
    gemm_bf16_wmma<<<dim3(VzP / BN, Mrows / BM), blk, 0, stream>>>(
        hb, wtep, nullptr, nullptr, (float*)d_out, Mrows, Vz, Dz, 0);
}
